// SelfAttention_5609227288840
// MI455X (gfx1250) — matrix operand order repaired and checked by reading, not compiled, not run on hardware
//
#include <hip/hip_runtime.h>
#include <hip/hip_bf16.h>
#include <math.h>

typedef __attribute__((ext_vector_type(16))) __bf16 v16bf;
typedef __attribute__((ext_vector_type(8)))  __bf16 v8bf;
typedef __attribute__((ext_vector_type(8)))  float  v8f;

#define TT 4096
#define CC 768
#define HH 12
#define DD 64
#define PROW 72  // padded LDS row stride (bf16 elems): 144B, breaks bank conflicts

__device__ __forceinline__ v16bf combine16(v8bf lo, v8bf hi) {
  v16bf r;
#pragma unroll
  for (int e = 0; e < 8; ++e) { r[e] = lo[e]; r[8 + e] = hi[e]; }
  return r;
}

__device__ __forceinline__ v8f wmma_bf16(v16bf a, v16bf b, v8f c) {
  return __builtin_amdgcn_wmma_f32_16x16x32_bf16(false, a, false, b, (short)0, c,
                                                 false, false);
}

// ---------------------------------------------------------------------------
// Kernel 1: fp32 -> bf16 conversions + layout transforms.
//   xbf[t][c] = bf16(x[t][c])
//   vt [c][t] = bf16(x[t][c])        (V transposed: P@V B-frags contiguous)
//   wkt[n][k] = bf16(Wk[k][n])       (W transposed: GEMM B-frags contiguous)
//   wqt[n][k] = bf16(Wq[k][n])
// ---------------------------------------------------------------------------
__global__ void __launch_bounds__(256)
convert_kernel(const float* __restrict__ x, const float* __restrict__ Wk,
               const float* __restrict__ Wq, __bf16* __restrict__ xbf,
               __bf16* __restrict__ vt, __bf16* __restrict__ wkt,
               __bf16* __restrict__ wqt) {
  int i = blockIdx.x * blockDim.x + threadIdx.x;
  if (i < TT * CC) {
    float xv = x[i];
    xbf[i] = (__bf16)xv;
    int t = i / CC;
    int c = i - t * CC;
    vt[(size_t)c * TT + t] = (__bf16)xv;
  }
  if (i < CC * CC) {
    int k = i / CC;
    int n = i - k * CC;
    wkt[(size_t)n * CC + k] = (__bf16)Wk[i];
    wqt[(size_t)n * CC + k] = (__bf16)Wq[i];
  }
}

// ---------------------------------------------------------------------------
// Kernel 2: projection GEMM via WMMA bf16.
//   out[h][t][d] = bf16( x[t][:] @ W[:, h*64+d] + b[h*64+d] )
// One 16x16 output tile per wave; K-loop over 768 in steps of 32.
// blockIdx.y selects K-projection (0) or Q-projection (1).
// ---------------------------------------------------------------------------
__global__ void __launch_bounds__(128)
proj_kernel(const __bf16* __restrict__ xbf, const __bf16* __restrict__ wkt,
            const __bf16* __restrict__ wqt, const float* __restrict__ bk,
            const float* __restrict__ bq, __bf16* __restrict__ kbf,
            __bf16* __restrict__ qbf) {
  const int warp = threadIdx.x >> 5;
  const int lane = threadIdx.x & 31;
  const int n_l  = lane & 15;
  const int hi   = lane >> 4;
  const int tile = blockIdx.x * 4 + warp;  // 0 .. 12287
  const int tt = tile / (CC / 16);         // 0 .. 255
  const int nt = tile - tt * (CC / 16);    // 0 .. 47

  const __bf16* wt  = (blockIdx.y == 0) ? wkt : wqt;
  const float*  bb  = (blockIdx.y == 0) ? bk : bq;
  __bf16*       obf = (blockIdx.y == 0) ? kbf : qbf;

  const __bf16* xrow = xbf + (size_t)(tt * 16 + n_l) * CC;  // A: row = lane&15
  const __bf16* wrow = wt + (size_t)(nt * 16 + n_l) * CC;   // B: col = lane&15

  v8f acc = {};
#pragma unroll 4
  for (int k0 = 0; k0 < CC; k0 += 32) {
    // A frag (16x32 bf16): element e -> K = (e>>3)*16 + hi*8 + (e&7)
    v8bf alo = *(const v8bf*)(xrow + k0 + hi * 8);
    v8bf ahi = *(const v8bf*)(xrow + k0 + 16 + hi * 8);
    v16bf af = combine16(alo, ahi);
    // B frag (32x16 bf16): element e -> K = (e>>3)*16 + hi*8 + (e&7)
    v16bf bf_ = combine16(*(const v8bf*)(wrow + k0 + hi * 8),
                          *(const v8bf*)(wrow + k0 + 16 + hi * 8));
    acc = wmma_bf16(af, bf_, acc);
  }

  const float bias = bb[nt * 16 + n_l];
  const int h = nt >> 2;
  const int d = (nt & 3) * 16 + n_l;
#pragma unroll
  for (int r = 0; r < 8; ++r) {  // C/D: row m = r + 8*hi
    int m = tt * 16 + r + 8 * hi;
    obf[((size_t)h * TT + m) * DD + d] = (__bf16)(acc[r] + bias);
  }
}

// ---------------------------------------------------------------------------
// Kernel 3: causal flash attention (reference computes K@Q^T, so Kp plays the
// "query" role, Qp the "key" role, V = raw x). One wave owns 16 output rows
// of one head; streams 64-key tiles with online softmax (amortizes the
// per-tile shfl reductions / rescale / LDS round-trip over 2x more keys).
// ---------------------------------------------------------------------------
__global__ void __launch_bounds__(128)
attn_kernel(const __bf16* __restrict__ kbf, const __bf16* __restrict__ qbf,
            const __bf16* __restrict__ vt, float* __restrict__ out) {
  __shared__ __bf16 plds[4][16 * PROW];  // per-wave P staging, padded rows

  const int warp = threadIdx.x >> 5;
  const int lane = threadIdx.x & 31;
  const int n_l  = lane & 15;
  const int hi   = lane >> 4;
  const int id = blockIdx.x * 4 + warp;  // 0 .. 3071
  const int h = id / (TT / 16);
  const int row16 = id - h * (TT / 16);
  const int i0 = row16 * 16;

  // A fragments: 16 K-projection rows x 64 channels (two 16x32 frags)
  const __bf16* krow = kbf + ((size_t)h * TT + i0 + n_l) * DD;
  v16bf a0 = combine16(*(const v8bf*)(krow + hi * 8),
                       *(const v8bf*)(krow + 16 + hi * 8));
  v16bf a1 = combine16(*(const v8bf*)(krow + 32 + hi * 8),
                       *(const v8bf*)(krow + 48 + hi * 8));

  v8f o[4] = {{0}, {0}, {0}, {0}};
  float mrow[8], lrow[8];
#pragma unroll
  for (int r = 0; r < 8; ++r) { mrow[r] = -INFINITY; lrow[r] = 0.f; }

  const int jtLast = (i0 + 15) / 64;
  for (int jt = 0; jt <= jtLast; ++jt) {
    const int j0 = jt * 64;

    // ---- S = Kp @ Qp^T : four 16x16 score tiles covering 64 keys ----
    v8f s[4];
#pragma unroll
    for (int c = 0; c < 4; ++c) {
      const __bf16* qr = qbf + ((size_t)h * TT + j0 + 16 * c + n_l) * DD;
      v16bf b0 = combine16(*(const v8bf*)(qr + hi * 8),
                           *(const v8bf*)(qr + 16 + hi * 8));
      v16bf b1 = combine16(*(const v8bf*)(qr + 32 + hi * 8),
                           *(const v8bf*)(qr + 48 + hi * 8));
      v8f sc = {};
      sc = wmma_bf16(a0, b0, sc);
      sc = wmma_bf16(a1, b1, sc);
      s[c] = sc;
    }

    // ---- causal mask (only the final tile crosses the diagonal) ----
    if (jt == jtLast) {
#pragma unroll
      for (int c = 0; c < 4; ++c) {
        const int jg = j0 + 16 * c + n_l;
#pragma unroll
        for (int r = 0; r < 8; ++r) {
          if (jg > i0 + r + 8 * hi) s[c][r] = -INFINITY;
        }
      }
    }

    // ---- online softmax; row stats shared across each 16-lane half ----
#pragma unroll
    for (int r = 0; r < 8; ++r) {
      float mx = fmaxf(fmaxf(s[0][r], s[1][r]), fmaxf(s[2][r], s[3][r]));
      mx = fmaxf(mx, __shfl_xor(mx, 1, 32));
      mx = fmaxf(mx, __shfl_xor(mx, 2, 32));
      mx = fmaxf(mx, __shfl_xor(mx, 4, 32));
      mx = fmaxf(mx, __shfl_xor(mx, 8, 32));
      const float mnew = fmaxf(mrow[r], mx);
      const float scale = __expf(mrow[r] - mnew);
      float p0 = __expf(s[0][r] - mnew);
      float p1 = __expf(s[1][r] - mnew);
      float p2 = __expf(s[2][r] - mnew);
      float p3 = __expf(s[3][r] - mnew);
      float ls = (p0 + p1) + (p2 + p3);
      ls += __shfl_xor(ls, 1, 32);
      ls += __shfl_xor(ls, 2, 32);
      ls += __shfl_xor(ls, 4, 32);
      ls += __shfl_xor(ls, 8, 32);
      lrow[r] = lrow[r] * scale + ls;
      mrow[r] = mnew;
      o[0][r] *= scale; o[1][r] *= scale; o[2][r] *= scale; o[3][r] *= scale;
      // stash P (C-layout) for the LDS transpose into A-layout
      __bf16* pr = &plds[warp][(r + 8 * hi) * PROW + n_l];
      pr[0]  = (__bf16)p0;
      pr[16] = (__bf16)p1;
      pr[32] = (__bf16)p2;
      pr[48] = (__bf16)p3;
    }

    // DS ops from one wave execute in order; wait for data + fence compiler.
    asm volatile("s_wait_dscnt 0" ::: "memory");
    const __bf16* prow = &plds[warp][n_l * PROW];
    v16bf pa0 = combine16(*(const v8bf*)(prow + hi * 16),
                          *(const v8bf*)(prow + 8 + hi * 16));
    v16bf pa1 = combine16(*(const v8bf*)(prow + 32 + hi * 16),
                          *(const v8bf*)(prow + 40 + hi * 16));

    // ---- O += P @ V ; V^T layout makes B-frags contiguous 32B loads ----
#pragma unroll
    for (int dt = 0; dt < 4; ++dt) {
      const __bf16* vrow =
          vt + ((size_t)(h * DD + dt * 16 + n_l)) * TT + j0 + hi * 16;
      v16bf bv0 = *(const v16bf*)vrow;
      v16bf bv1 = *(const v16bf*)(vrow + 32);
      if (jt < jtLast) __builtin_prefetch(vrow + 64, 0, 3);  // next key tile
      o[dt] = wmma_bf16(pa0, bv0, o[dt]);
      o[dt] = wmma_bf16(pa1, bv1, o[dt]);
    }
  }

  // ---- epilogue: normalize and write fp32 output [t][h*64+d] ----
#pragma unroll
  for (int dt = 0; dt < 4; ++dt) {
#pragma unroll
    for (int r = 0; r < 8; ++r) {
      const int m = i0 + r + 8 * hi;
      out[(size_t)m * CC + h * DD + dt * 16 + n_l] = o[dt][r] / lrow[r];
    }
  }
}

// ---------------------------------------------------------------------------
extern "C" void kernel_launch(void* const* d_in, const int* in_sizes, int n_in,
                              void* d_out, int out_size, void* d_ws,
                              size_t ws_size, hipStream_t stream) {
  const float* x  = (const float*)d_in[0];
  const float* Wk = (const float*)d_in[1];
  const float* bk = (const float*)d_in[2];
  const float* Wq = (const float*)d_in[3];
  const float* bq = (const float*)d_in[4];
  float* out = (float*)d_out;

  // Workspace layout (bf16 elements): xbf | kbf | qbf | vt | wkt | wqt
  __bf16* xbf = (__bf16*)d_ws;
  __bf16* kbf = xbf + (size_t)TT * CC;
  __bf16* qbf = kbf + (size_t)TT * CC;
  __bf16* vtp = qbf + (size_t)TT * CC;
  __bf16* wkt = vtp + (size_t)TT * CC;
  __bf16* wqt = wkt + (size_t)CC * CC;

  convert_kernel<<<(TT * CC + 255) / 256, 256, 0, stream>>>(x, Wk, Wq, xbf,
                                                            vtp, wkt, wqt);

  // 256 * 48 = 12288 wave-tiles, 4 waves per block, y=0 -> K, y=1 -> Q
  proj_kernel<<<dim3(12288 / 4, 2), 128, 0, stream>>>(xbf, wkt, wqt, bk, bq,
                                                      kbf, qbf);

  // 12 heads * 256 row-tiles = 3072 wave-tiles, 4 waves per block
  attn_kernel<<<3072 / 4, 128, 0, stream>>>(kbf, qbf, vtp, out);
}
